// MultiScaleTrendMixing_54417235640435
// MI455X (gfx1250) — matrix operand order repaired and checked by reading, not compiled, not run on hardware
//
#include <hip/hip_runtime.h>
#include <math.h>
#include <stdint.h>

#define BB 32
#define SS 128
#define HH 128
#define NTOT (BB * SS * HH)   // 524288 elements per [B,S,H] tensor

#define USE_ASYNC_LDS 1

#if defined(__has_builtin)
#if __has_builtin(__builtin_amdgcn_tensor_load_to_lds) && __has_builtin(__builtin_amdgcn_s_wait_tensorcnt)
#define USE_TDM 1
#endif
#endif

typedef __attribute__((ext_vector_type(16))) _Float16 v16h;
typedef __attribute__((ext_vector_type(8)))  _Float16 v8h;
typedef __attribute__((ext_vector_type(8)))  float    v8f;
typedef __attribute__((ext_vector_type(4)))  unsigned int u32x4;
typedef __attribute__((ext_vector_type(8)))  int      i32x8;
typedef __attribute__((ext_vector_type(4)))  int      i32x4;

// Branchless tanh-form GELU.  gfx1250 has V_TANH_F32 (TRANS pipe).
__device__ __forceinline__ float gelu_fast(float x) {
    const float x2 = x * x;
    const float v  = x * __builtin_fmaf(0.0356774081f, x2, 0.7978845608f);
#if defined(__has_builtin) && __has_builtin(__builtin_amdgcn_tanhf)
    const float th = __builtin_amdgcn_tanhf(v);
    const float hx = 0.5f * x;
    return __builtin_fmaf(hx, th, hx);          // 0.5x(1+tanh(v))
#else
    const float t  = __expf(-2.0f * v);         // sigmoid form fallback
    return x * __builtin_amdgcn_rcpf(1.0f + t);
#endif
}

// Copy 32 KB contiguous (16384 f16) global -> LDS with 256 threads
// via async global->LDS loads (ASYNCcnt path).
__device__ __forceinline__ void stage32k_async(const _Float16* __restrict__ gsrc,
                                               _Float16* lbase, int tid) {
#if USE_ASYNC_LDS
    uint64_t g = (uint64_t)(uintptr_t)gsrc + (uint32_t)tid * 16u;
    uint32_t l = (uint32_t)(uintptr_t)lbase + (uint32_t)tid * 16u;  // LDS byte addr = addr[31:0]
#pragma unroll
    for (int c = 0; c < 8; ++c) {
        asm volatile("global_load_async_to_lds_b128 %0, %1, off"
                     :
                     : "v"(l + (uint32_t)(c * 4096)), "v"(g + (uint64_t)(c * 4096))
                     : "memory");
    }
#else
    const v8h* gp = (const v8h*)gsrc + tid;
    v8h* lp = (v8h*)lbase + tid;
#pragma unroll
    for (int c = 0; c < 8; ++c) lp[c * 256] = gp[c * 256];
#endif
}

__device__ __forceinline__ void wait_async_lds() {
#if USE_ASYNC_LDS
    asm volatile("s_wait_asynccnt 0" ::: "memory");
#endif
}

#ifdef USE_TDM
// One-shot TDM: 32 KB contiguous tile (4096 x 8B) global -> LDS.
// D# per CDNA5 ISA: count=1, type=2("image"),
// data_size=3 (8B), tensor_dim0 = tile_dim0 = 4096, tile_dim1 = 1.
__device__ __forceinline__ void tdm_stage32k_issue(const _Float16* gsrc, _Float16* lbase) {
    const uint64_t ga = (uint64_t)(uintptr_t)gsrc;
    const uint32_t la = (uint32_t)(uintptr_t)lbase;
    u32x4 g0;
    g0[0] = 1u;                                                 // count=1, user descriptor
    g0[1] = la;                                                 // lds_addr
    g0[2] = (uint32_t)ga;                                       // global_addr[31:0]
    g0[3] = (uint32_t)((ga >> 32) & 0x1FFFFFFu) | (2u << 30);   // global_addr[56:32] | type=2
    i32x8 g1;
    g1[0] = (int)(3u << 16);        // workgroup_mask=0, data_size=3 (8 bytes)
    g1[1] = (int)(4096u << 16);     // tensor_dim0 low16 @ bits 63:48
    g1[2] = (int)(1u << 16);        // tensor_dim0 hi16 = 0, tensor_dim1 = 1 (low16)
    g1[3] = (int)(4096u << 16);     // tensor_dim1 hi16 = 0, tile_dim0 = 4096
    g1[4] = 1;                      // tile_dim1 = 1, tile_dim2 = 0
    g1[5] = 4096;                   // tensor_dim0_stride low32 (data_size units)
    g1[6] = 0;                      // stride hi / tensor_dim1_stride lo
    g1[7] = 0;
    const i32x4 z4 = {0, 0, 0, 0};
    const i32x8 z8 = {0, 0, 0, 0, 0, 0, 0, 0};
    __builtin_amdgcn_tensor_load_to_lds(g0, g1, z4, z4, z8, 0);
}
#endif

// A-fragment (16x32 f16) for v_wmma_f32_16x16x32_f16 from row-major [rows][128] f16.
// lane: m = row, element e: k = kk + 16*(lane>>4) + e
__device__ __forceinline__ v16h load_a_frag(const _Float16* __restrict__ A,
                                            int row, int kk, int lane) {
    const int khalf = (lane >> 4) * 16;
    const _Float16* p = A + (size_t)row * 128 + kk + khalf;
    v8h lo = *(const v8h*)(p);
    v8h hi = *(const v8h*)(p + 8);
    v16h a;
#pragma unroll
    for (int e = 0; e < 8; ++e) { a[e] = lo[e]; a[8 + e] = hi[e]; }
    return a;
}

// ---------------------------------------------------------------------------
// mix kernel: Y[bh,t] = sum_s A16[bh,s] * W1t[l,j][t,s] ; then
//             res[bh] = sum_t gelu(Y + b1[l,j,t]) * W2[l,j,t]
//             X16[b,j,h] = f16( base[b,j,h] + res + b2[l,j] )
// grid = (128 j, 32 row-blocks) x 256 threads (8 waves x 16 rows)
// W1t[l,j] tile staged into LDS via the Tensor Data Mover.
// ---------------------------------------------------------------------------
__global__ __launch_bounds__(256) void mix_gemm_kernel(
    const _Float16* __restrict__ A16, const _Float16* __restrict__ W1t,
    const float* __restrict__ b1, const float* __restrict__ W2,
    const float* __restrict__ b2, const float* __restrict__ base,
    _Float16* __restrict__ X16, int layer)
{
    __shared__ __align__(16) _Float16 Bt[SS * SS];  // [t][s], 32 KB

    const int j    = blockIdx.x;
    const int tid  = threadIdx.x;
    const int wave = tid >> 5, lane = tid & 31;

    const _Float16* __restrict__ wsrc = W1t + (((size_t)layer * SS) + j) * SS * SS;
#ifdef USE_TDM
    if (wave == 0) tdm_stage32k_issue(wsrc, Bt);
#else
    stage32k_async(wsrc, Bt, tid);
#endif

    const int m0   = blockIdx.y * 128 + wave * 16;
    const int mrow = m0 + (lane & 15);

    v16h afr[4];
#pragma unroll
    for (int kt = 0; kt < 4; ++kt) afr[kt] = load_a_frag(A16, mrow, kt * 32, lane);

#ifdef USE_TDM
    if (wave == 0) __builtin_amdgcn_s_wait_tensorcnt(0);
#else
    wait_async_lds();
#endif
    __syncthreads();

    float acc[8];
#pragma unroll
    for (int r = 0; r < 8; ++r) acc[r] = 0.0f;

    const float* __restrict__ b1j = b1 + (((size_t)layer * SS) + j) * SS;
    const float* __restrict__ W2j = W2 + (((size_t)layer * SS) + j) * SS;
    const int k0 = (lane >> 4) * 16;

#pragma unroll
    for (int nt = 0; nt < 8; ++nt) {
        const int nb = nt * 16 + (lane & 15);
        // preload all 4 K-step B fragments so dscnt waits can cascade
        v16h bfr[4];
#pragma unroll
        for (int kt = 0; kt < 4; ++kt)
            bfr[kt] = *(const v16h*)(&Bt[nb * SS + kt * 32 + k0]);
        v8f c = {};
#pragma unroll
        for (int kt = 0; kt < 4; ++kt)
            c = __builtin_amdgcn_wmma_f32_16x16x32_f16(
                    false, afr[kt], false, bfr[kt], (short)0, c, false, false);
        const float b1v = b1j[nb];
        const float w2v = W2j[nb];
#pragma unroll
        for (int r = 0; r < 8; ++r)
            acc[r] = __builtin_fmaf(gelu_fast(c[r] + b1v), w2v, acc[r]);
    }

    // butterfly reduce over 16-lane column-class groups
#pragma unroll
    for (int off = 1; off < 16; off <<= 1) {
#pragma unroll
        for (int r = 0; r < 8; ++r) acc[r] += __shfl_xor(acc[r], off, 32);
    }

    const int lr = lane & 15;
    if (lr < 8) {
        const int row = m0 + (lane >> 4) * 8 + lr;  // bh
        const int b = row >> 7, h = row & 127;
        const size_t o = ((size_t)b * SS + j) * HH + h;
        X16[o] = (_Float16)(base[o] + acc[lr] + b2[layer * SS + j]);
    }
}

// ---------------------------------------------------------------------------
// Fused MLP: out = gelu(X @ Wm1 + bm1) @ Wm2 + bm2 ; also emits the packed
// transposed f16 activations A16p[(b*H+h)][s] for the next mix layer.
// Weight staging uses the async global->LDS path.
// ---------------------------------------------------------------------------
__global__ __launch_bounds__(256) void mlp_kernel(
    const _Float16* __restrict__ X16,
    const _Float16* __restrict__ Wm1t, const _Float16* __restrict__ Wm2t,
    const float* __restrict__ bm1, const float* __restrict__ bm2,
    float* __restrict__ out, _Float16* __restrict__ A16p)
{
    __shared__ __align__(16) _Float16 Wt[HH * HH];       // staged weight^T (32 KB)
    __shared__ __align__(16) _Float16 Hd[8 * 16 * HH];   // per-wave hidden (32 KB)

    const int tid  = threadIdx.x;
    const int wave = tid >> 5, lane = tid & 31;
    const int m0   = blockIdx.x * 128 + wave * 16;
    const int mrow = m0 + (lane & 15);
    const int k0   = (lane >> 4) * 16;

    stage32k_async(Wm1t, Wt, tid);

    v16h afr[4];
#pragma unroll
    for (int kt = 0; kt < 4; ++kt) afr[kt] = load_a_frag(X16, mrow, kt * 32, lane);

    wait_async_lds();
    __syncthreads();

    // GEMM 1 + bias + GELU -> f16 hidden in own-wave LDS region
#pragma unroll
    for (int nt = 0; nt < 8; ++nt) {
        const int nb = nt * 16 + (lane & 15);
        v16h bfr[4];
#pragma unroll
        for (int kt = 0; kt < 4; ++kt)
            bfr[kt] = *(const v16h*)(&Wt[nb * HH + kt * 32 + k0]);
        v8f c = {};
#pragma unroll
        for (int kt = 0; kt < 4; ++kt)
            c = __builtin_amdgcn_wmma_f32_16x16x32_f16(
                    false, afr[kt], false, bfr[kt], (short)0, c, false, false);
        const float bm1v = bm1[nb];
#pragma unroll
        for (int r = 0; r < 8; ++r) {
            const int mloc = r + 8 * (lane >> 4);
            Hd[wave * (16 * HH) + mloc * HH + nb] = (_Float16)gelu_fast(c[r] + bm1v);
        }
    }
    __syncthreads();                  // everyone done reading Wm1^T
    stage32k_async(Wm2t, Wt, tid);    // overwrite with Wm2^T
    wait_async_lds();
    __syncthreads();

    // GEMM 2: A fragments from own hidden LDS region
    const _Float16* hb = &Hd[wave * (16 * HH)];
#pragma unroll
    for (int kt = 0; kt < 4; ++kt) afr[kt] = load_a_frag(hb, lane & 15, kt * 32, lane);

#pragma unroll
    for (int nt = 0; nt < 8; ++nt) {
        const int nb = nt * 16 + (lane & 15);
        v16h bfr[4];
#pragma unroll
        for (int kt = 0; kt < 4; ++kt)
            bfr[kt] = *(const v16h*)(&Wt[nb * HH + kt * 32 + k0]);
        v8f c = {};
#pragma unroll
        for (int kt = 0; kt < 4; ++kt)
            c = __builtin_amdgcn_wmma_f32_16x16x32_f16(
                    false, afr[kt], false, bfr[kt], (short)0, c, false, false);
        const float bm2v = bm2[nb];
#pragma unroll
        for (int r = 0; r < 8; ++r) {
            const int row = m0 + r + 8 * (lane >> 4);   // (b*S+s) flat row
            const float val = c[r] + bm2v;
            out[(size_t)row * HH + nb] = val;
            const int b = row >> 7, s = row & 127;
            A16p[((size_t)(b * HH + nb)) * SS + s] = (_Float16)val;  // packed for next layer
        }
    }
}

// Pre-transpose + f16-convert W1:  W1t[l][j][t][s] = f16(W1[l][j][s][t])
__global__ void prep_w1_kernel(const float* __restrict__ W1, _Float16* __restrict__ W1t) {
    const unsigned idx = blockIdx.x * 256 + threadIdx.x;   // over 2*S^3
    if (idx >= 2u * SS * SS * SS) return;
    const int s = idx & 127;
    const int t = (idx >> 7) & 127;
    const unsigned lj = idx >> 14;
    W1t[idx] = (_Float16)W1[((size_t)lj << 14) + (unsigned)s * SS + t];
}

// Pre-transpose + f16-convert MLP weights
__global__ void prep_wm_kernel(const float* __restrict__ Wm1, const float* __restrict__ Wm2,
                               _Float16* __restrict__ Wm1t, _Float16* __restrict__ Wm2t) {
    const int idx = blockIdx.x * 256 + threadIdx.x;  // 16384
    const int n = idx >> 7, k = idx & 127;
    Wm1t[idx] = (_Float16)Wm1[k * HH + n];
    Wm2t[idx] = (_Float16)Wm2[k * HH + n];
}

// initial pack: A16[(b*H+h)][s] = f16(src[b][s][h])
__global__ void pack_bhs_kernel(const float* __restrict__ src, _Float16* __restrict__ dst) {
    const int idx = blockIdx.x * 256 + threadIdx.x;
    if (idx >= NTOT) return;
    const int h = idx & 127;
    const int s = (idx >> 7) & 127;
    const int b = idx >> 14;
    dst[((size_t)(b * HH + h)) * SS + s] = (_Float16)src[idx];
}

extern "C" void kernel_launch(void* const* d_in, const int* in_sizes, int n_in,
                              void* d_out, int out_size, void* d_ws, size_t ws_size,
                              hipStream_t stream) {
    (void)in_sizes; (void)n_in; (void)out_size; (void)ws_size;

    const float* trend0 = (const float*)d_in[0];
    const float* trend1 = (const float*)d_in[1];
    const float* trend2 = (const float*)d_in[2];
    const float* W1     = (const float*)d_in[3];
    const float* b1     = (const float*)d_in[4];
    const float* W2     = (const float*)d_in[5];
    const float* b2     = (const float*)d_in[6];
    const float* Wm1    = (const float*)d_in[7];
    const float* bm1    = (const float*)d_in[8];
    const float* Wm2    = (const float*)d_in[9];
    const float* bm2    = (const float*)d_in[10];

    float* out_M1 = (float*)d_out;              // outs reversed: [M1, M0, trend2]
    float* out_M0 = out_M1 + NTOT;
    float* out_T2 = out_M1 + 2 * (size_t)NTOT;

    char* ws = (char*)d_ws;
    _Float16* A16  = (_Float16*)ws;                       // 1 MB    packed activations [bh][s]
    _Float16* X16  = (_Float16*)(ws + (1u << 20));        // 1 MB    mixed intermediate f16 [b,s,h]
    _Float16* W1t  = (_Float16*)(ws + (2u << 20));        // 8 MB    W1 transposed f16 [l,j,t,s]
    _Float16* Wm1t = (_Float16*)(ws + (10u << 20));       // 32 KB
    _Float16* Wm2t = (_Float16*)(ws + (10u << 20) + 32768);

    const dim3 blk(256);
    const dim3 gelem((NTOT + 255) / 256);
    const dim3 gw1((2u * SS * SS * SS + 255) / 256);
    const dim3 gmix(SS, 32);
    const dim3 gmlp(32);

    prep_w1_kernel<<<gw1, blk, 0, stream>>>(W1, W1t);
    prep_wm_kernel<<<dim3(64), blk, 0, stream>>>(Wm1, Wm2, Wm1t, Wm2t);
    (void)hipMemcpyAsync(out_T2, trend2, NTOT * sizeof(float), hipMemcpyDeviceToDevice, stream);
    pack_bhs_kernel<<<gelem, blk, 0, stream>>>(trend2, A16);

    // ---- i = 0: out_low = trend2, residual base = trend1 ----
    mix_gemm_kernel<<<gmix, blk, 0, stream>>>(A16, W1t, b1, W2, b2, trend1, X16, 0);
    mlp_kernel<<<gmlp, blk, 0, stream>>>(X16, Wm1t, Wm2t, bm1, bm2, out_M0, A16);

    // ---- i = 1: out_low = M0, residual base = trend0 ----
    mix_gemm_kernel<<<gmix, blk, 0, stream>>>(A16, W1t, b1, W2, b2, trend0, X16, 1);
    mlp_kernel<<<gmlp, blk, 0, stream>>>(X16, Wm1t, Wm2t, bm1, bm2, out_M1, A16);
}
